// DTransformer_33105607917999
// MI455X (gfx1250) — matrix operand order repaired and checked by reading, not compiled, not run on hardware
//
#include <hip/hip_runtime.h>
#include <hip/hip_bf16.h>

// ---------------------------------------------------------------------------
// Types for CDNA5 WMMA (wave32): v_wmma_f32_16x16x32_bf16
// ---------------------------------------------------------------------------
typedef __attribute__((ext_vector_type(16))) __bf16 bf16x16;
typedef __attribute__((ext_vector_type(8)))  float  v8f;

#define WMMA_BF16(A, Bv, C) \
  __builtin_amdgcn_wmma_f32_16x16x32_bf16(false, (A), false, (Bv), (short)0, (C), false, false)

__device__ __forceinline__ unsigned short f2bf(float f) {
  unsigned int u = __float_as_uint(f);
  // round-to-nearest-even bf16
  return (unsigned short)((u + 0x7FFFu + ((u >> 16) & 1u)) >> 16);
}

// Build a 16-element bf16 fragment from two contiguous 8-element (16B) chunks.
__device__ __forceinline__ bf16x16 frag2(const unsigned short* p0, const unsigned short* p1) {
  union { uint4 q[2]; bf16x16 v; } u;
  u.q[0] = *reinterpret_cast<const uint4*>(p0);
  u.q[1] = *reinterpret_cast<const uint4*>(p1);
  return u.v;
}

// ---------------------------------------------------------------------------
// CDNA5 async global->LDS DMA (ASYNCcnt path): copies 16B per active lane
// directly into LDS without a VGPR round trip or LOADcnt stall.
// LDS address = low 32 bits of the generic pointer (flat->LDS aperture keeps
// addr[31:0] as the workgroup-relative LDS offset).
// ---------------------------------------------------------------------------
__device__ __forceinline__ void async_g2l_b128(const void* gptr, void* lptr) {
  unsigned lds_off = (unsigned)(reinterpret_cast<uintptr_t>(lptr));
  asm volatile("global_load_async_to_lds_b128 %0, %1, off"
               :: "v"(lds_off), "v"(gptr)
               : "memory");
}

__device__ __forceinline__ void wait_async0() {
#if __has_builtin(__builtin_amdgcn_s_wait_asynccnt)
  __builtin_amdgcn_s_wait_asynccnt(0);
#else
  asm volatile("s_wait_asynccnt 0x0" ::: "memory");
#endif
}

// ---------------------------------------------------------------------------
// fp32 -> bf16 conversion (elementwise, 4 per thread)
// ---------------------------------------------------------------------------
__global__ __launch_bounds__(256) void cvt_bf16(const float* __restrict__ s,
                                                unsigned short* __restrict__ d, int n) {
  int i = (blockIdx.x * 256 + threadIdx.x) * 4;
  if (i + 3 < n) {
    float4 f = *reinterpret_cast<const float4*>(&s[i]);
    d[i + 0] = f2bf(f.x);
    d[i + 1] = f2bf(f.y);
    d[i + 2] = f2bf(f.z);
    d[i + 3] = f2bf(f.w);
  }
}

// ---------------------------------------------------------------------------
// Tiled WMMA GEMM:  Y[m,n] = sum_k X[m,k] * W[n,k] + bias[n]   (NT layout)
// Block tile 128x128, K-step 32. 8 waves, each 32x64 of output.
// Double-buffered LDS staging via async global->LDS DMA: the next K-tile's
// copies are in flight while the current tile feeds 8 WMMAs per wave.
// mode 0: store bf16 into per-head (b,h,t,dk) layout       (Q / K projection)
// mode 1: store bf16 into transposed (b,h,dk,t) layout     (V projection)
// mode 2: store f32 row-major (M,N)                        (O projection)
// ---------------------------------------------------------------------------
__global__ __launch_bounds__(256) void gemm_wmma(const unsigned short* __restrict__ X,
                                                 const unsigned short* __restrict__ W,
                                                 const float* __restrict__ bias,
                                                 void* __restrict__ out,
                                                 int M, int N, int K, int mode) {
  const int tid = threadIdx.x;
  const int wave = tid >> 5, lane = tid & 31;
  const int wm = wave >> 1, wn = wave & 1;        // 4x2 wave grid
  const int lh = lane >> 4, lo = lane & 15;
  const int bm = blockIdx.x, bn = blockIdx.y;

  __shared__ __align__(16) unsigned short As[2][128 * 32];  // 2 x 8 KB
  __shared__ __align__(16) unsigned short Bs[2][128 * 32];  // 2 x 8 KB

  const int row = tid >> 1;             // staging row (2 chunks/thread/array)
  const int c8a = (tid & 1) * 8;        // chunk 0 / 1 within row (16B each)

  const v8f vzero = {0.f, 0.f, 0.f, 0.f, 0.f, 0.f, 0.f, 0.f};
  v8f acc[2][4];
  for (int sm = 0; sm < 2; ++sm)
    for (int sn = 0; sn < 4; ++sn) acc[sm][sn] = vzero;

  // Issue async DMA for one 128x32 K-tile of X and W into LDS buffer `buf`.
  auto stage = [&](int buf, int kk) {
#pragma unroll
    for (int i = 0; i < 2; ++i) {
      int c8 = c8a + i * 16;
      async_g2l_b128(&X[(size_t)(bm * 128 + row) * K + kk + c8], &As[buf][row * 32 + c8]);
      async_g2l_b128(&W[(size_t)(bn * 128 + row) * K + kk + c8], &Bs[buf][row * 32 + c8]);
    }
  };

  stage(0, 0);
  wait_async0();
  __syncthreads();

  int cur = 0;
  for (int kk = 0; kk < K; kk += 32) {
    if (kk + 32 < K) {
      stage(cur ^ 1, kk + 32);  // overlap next tile's DMA with compute
      __builtin_prefetch(&X[(size_t)(bm * 128 + row) * K + kk + 64], 0, 1);
      __builtin_prefetch(&W[(size_t)(bn * 128 + row) * K + kk + 64], 0, 1);
    }

#pragma unroll
    for (int sm = 0; sm < 2; ++sm) {
      int m = wm * 32 + sm * 16 + lo;
      bf16x16 a = frag2(&As[cur][m * 32 + 8 * lh], &As[cur][m * 32 + 16 + 8 * lh]);
#pragma unroll
      for (int sn = 0; sn < 4; ++sn) {
        int n = wn * 64 + sn * 16 + lo;
        bf16x16 b = frag2(&Bs[cur][n * 32 + 8 * lh], &Bs[cur][n * 32 + 16 + 8 * lh]);
        acc[sm][sn] = WMMA_BF16(a, b, acc[sm][sn]);
      }
    }

    wait_async0();      // next-tile DMA landed
    __syncthreads();    // all waves done reading `cur`; safe to flip
    cur ^= 1;
  }

  // Epilogue: D layout is VGPR r -> row (r + 8*half), lane lo -> col.
#pragma unroll
  for (int sm = 0; sm < 2; ++sm) {
#pragma unroll
    for (int sn = 0; sn < 4; ++sn) {
#pragma unroll
      for (int r = 0; r < 8; ++r) {
        int mg = bm * 128 + wm * 32 + sm * 16 + r + 8 * lh;  // token row (b*T + t)
        int ng = bn * 128 + wn * 64 + sn * 16 + lo;          // feature (h*dk + d)
        float val = acc[sm][sn][r] + bias[ng];
        if (mode == 0) {          // (b,h,t,dk) bf16
          size_t addr = (((size_t)((mg >> 10) * 16 + (ng >> 6)) * 1024 + (mg & 1023)) << 6) + (ng & 63);
          ((unsigned short*)out)[addr] = f2bf(val);
        } else if (mode == 1) {   // (b,h,dk,t) bf16  (V transposed for p@V B-fragments)
          size_t addr = (((size_t)((mg >> 10) * 16 + (ng >> 6)) * 64 + (ng & 63)) << 10) + (mg & 1023);
          ((unsigned short*)out)[addr] = f2bf(val);
        } else {                  // f32 row-major
          ((float*)out)[(size_t)mg * N + ng] = val;
        }
      }
    }
  }
}

// ---------------------------------------------------------------------------
// Attention core: one workgroup per (query-tile of 16 rows, b*H + h).
// Phase 1: scores(16x1024) = Q_tile @ K^T / 8 via WMMA (8 waves split N-tiles)
// Phase 2: masked softmax -> cumsum -> decay(total_effect) -> rescale
//          -> masked softmax -> maxout scale; p stored bf16 in LDS.
//          (16 rows x 16 segments of 64 cols; one thread per segment.)
// Phase 3: out(16x64) = p @ V via WMMA (4 waves), scatter to concat (B,T,D).
// ---------------------------------------------------------------------------
__global__ __launch_bounds__(256) void attn_core(const unsigned short* __restrict__ Qb,
                                                 const unsigned short* __restrict__ Kb,
                                                 const unsigned short* __restrict__ Vt,
                                                 const float* __restrict__ gam,
                                                 unsigned short* __restrict__ concat) {
  const int T = 1024, DK = 64;
  const int qt = blockIdx.x;
  const int bh = blockIdx.y;
  const int b = bh >> 4, h = bh & 15;
  const int tid = threadIdx.x;
  const int wave = tid >> 5, lane = tid & 31;
  const int lh = lane >> 4, lo = lane & 15;

  __shared__ __align__(16) float sc[16 * 1024];           // 64 KB scores
  __shared__ __align__(16) unsigned short pb[16 * 1024];  // 32 KB p (bf16)
  __shared__ __align__(16) unsigned short qtile[16 * 64]; // 2 KB
  __shared__ float red[16][16];
  __shared__ float spre[16][16];
  __shared__ float rmax[16];
  __shared__ float rsum[16];

  const size_t headQK = (size_t)bh * T * DK;  // (B*H, T, DK) layout

  if (tid < 128) {  // stage Q tile (16 x 64 bf16) via async DMA
    int qrow = tid >> 3, c8 = (tid & 7) * 8;
    async_g2l_b128(&Qb[headQK + (size_t)(qt * 16 + qrow) * DK + c8], &qtile[qrow * 64 + c8]);
  }
  wait_async0();
  __syncthreads();

  // ---- Phase 1: scores ----
  for (int nt = wave; nt < 64; nt += 8) {
    v8f acc = {0.f, 0.f, 0.f, 0.f, 0.f, 0.f, 0.f, 0.f};
#pragma unroll
    for (int ks = 0; ks < 2; ++ks) {
      int k = ks * 32;
      bf16x16 a = frag2(&qtile[lo * 64 + k + 8 * lh], &qtile[lo * 64 + k + 16 + 8 * lh]);
      const unsigned short* kp = &Kb[headQK + (size_t)(nt * 16 + lo) * DK + k + 8 * lh];
      bf16x16 bf = frag2(kp, kp + 16);
      acc = WMMA_BF16(a, bf, acc);
    }
#pragma unroll
    for (int r = 0; r < 8; ++r)
      sc[(r + 8 * lh) * 1024 + nt * 16 + lo] = acc[r] * 0.125f;  // 1/sqrt(64)
  }
  __syncthreads();

  // ---- Phase 2: row-wise pipeline ----
  const int r = tid >> 4, s = tid & 15;
  const int i = qt * 16 + r;         // global query index
  const int base = r * 1024 + s * 64;
  const float neg = -1e32f;

  // softmax #1 stats
  float mx = neg;
  for (int c = 0; c < 64; ++c) { int j = s * 64 + c; float v = (j < i) ? sc[base + c] : neg; mx = fmaxf(mx, v); }
  red[r][s] = mx; __syncthreads();
  if (s == 0) { float m = red[r][0]; for (int k = 1; k < 16; ++k) m = fmaxf(m, red[r][k]); rmax[r] = m; }
  __syncthreads();
  float M1 = rmax[r];
  float sum = 0.f;
  for (int c = 0; c < 64; ++c) { int j = s * 64 + c; float v = (j < i) ? sc[base + c] : neg; sum += __expf(v - M1); }
  red[r][s] = sum; __syncthreads();
  if (s == 0) { float tot = 0.f; for (int k = 0; k < 16; ++k) { spre[r][k] = tot; tot += red[r][k]; } rsum[r] = tot; }
  __syncthreads();

  // cumsum + gamma time-effect rescale
  float inv1 = 1.0f / rsum[r];
  float g = -fabsf(gam[h]);
  float run = spre[r][s];
  for (int c = 0; c < 64; ++c) {
    int j = s * 64 + c;
    float s0 = sc[base + c];
    float v = (j < i) ? s0 : neg;
    run += __expf(v - M1);
    float distcum = run * inv1;                               // disttotal == 1
    float pos = fabsf((float)(i - j));
    float dist = sqrtf(fmaxf((1.0f - distcum) * pos, 0.0f));
    float te = fminf(fmaxf(__expf(dist * g), 1e-5f), 1e5f);
    sc[base + c] = (j < i) ? (s0 * te) : neg;
  }
  __syncthreads();

  // softmax #2 + maxout rescale
  mx = neg;
  for (int c = 0; c < 64; ++c) mx = fmaxf(mx, sc[base + c]);
  red[r][s] = mx; __syncthreads();
  if (s == 0) { float m = red[r][0]; for (int k = 1; k < 16; ++k) m = fmaxf(m, red[r][k]); rmax[r] = m; }
  __syncthreads();
  float M2 = rmax[r];
  sum = 0.f;
  for (int c = 0; c < 64; ++c) sum += __expf(sc[base + c] - M2);
  red[r][s] = sum; __syncthreads();
  if (s == 0) { float tot = 0.f; for (int k = 0; k < 16; ++k) tot += red[r][k]; rsum[r] = tot; }
  __syncthreads();
  float S2 = rsum[r];
  float inv2 = 1.0f / S2;
  float scale = fminf(S2, 5.0f);   // min(1/max(p), 5) with max(p) = 1/S2
  for (int c = 0; c < 64; ++c) {
    int j = s * 64 + c;
    float p = (j < i) ? __expf(sc[base + c] - M2) * inv2 * scale : 0.f;
    pb[base + c] = f2bf(p);
  }
  __syncthreads();

  // ---- Phase 3: out = p @ V ----
  if (wave < 4) {
    v8f acc = {0.f, 0.f, 0.f, 0.f, 0.f, 0.f, 0.f, 0.f};
    const size_t headV = (size_t)bh * DK * T;   // (B*H, DK, T) layout
    int d = wave * 16 + lo;
    for (int kk = 0; kk < 32; ++kk) {
      int k = kk * 32;
      bf16x16 a = frag2(&pb[lo * 1024 + k + 8 * lh], &pb[lo * 1024 + k + 16 + 8 * lh]);
      const unsigned short* vp = &Vt[headV + (size_t)d * T + k + 8 * lh];
      bf16x16 bf = frag2(vp, vp + 16);
      acc = WMMA_BF16(a, bf, acc);
    }
#pragma unroll
    for (int rr = 0; rr < 8; ++rr) {
      int t = qt * 16 + rr + 8 * lh;
      int col = h * 64 + wave * 16 + lo;
      concat[((size_t)b * 1024 + t) * 1024 + col] = f2bf(acc[rr]);
    }
  }
}

// ---------------------------------------------------------------------------
// Residual + LayerNorm:  out = LN(query + attn) * ln_g + ln_b
// ---------------------------------------------------------------------------
__global__ __launch_bounds__(256) void resid_ln(const float* __restrict__ q,
                                                const float* __restrict__ a,
                                                const float* __restrict__ g,
                                                const float* __restrict__ bb,
                                                float* __restrict__ out) {
  int row = blockIdx.x, t = threadIdx.x;
  __shared__ float s1[256], s2[256];
  size_t base = (size_t)row * 1024;
  float x[4], sum = 0.f, sq = 0.f;
#pragma unroll
  for (int k = 0; k < 4; ++k) {
    int c = t + k * 256;
    float v = q[base + c] + a[base + c];
    x[k] = v; sum += v; sq += v * v;
  }
  s1[t] = sum; s2[t] = sq; __syncthreads();
  for (int st = 128; st > 0; st >>= 1) {
    if (t < st) { s1[t] += s1[t + st]; s2[t] += s2[t + st]; }
    __syncthreads();
  }
  float mu = s1[0] * (1.0f / 1024.0f);
  float var = s2[0] * (1.0f / 1024.0f) - mu * mu;
  float rstd = rsqrtf(var + 1e-5f);
#pragma unroll
  for (int k = 0; k < 4; ++k) {
    int c = t + k * 256;
    out[base + c] = (x[k] - mu) * rstd * g[c] + bb[c];
  }
}

// ---------------------------------------------------------------------------
// Host-side launcher
// ---------------------------------------------------------------------------
extern "C" void kernel_launch(void* const* d_in, const int* in_sizes, int n_in,
                              void* d_out, int out_size, void* d_ws, size_t ws_size,
                              hipStream_t stream) {
  (void)in_sizes; (void)n_in; (void)out_size; (void)ws_size;
  const float* query  = (const float*)d_in[0];
  const float* key    = (const float*)d_in[1];
  const float* values = (const float*)d_in[2];
  const float* Wq     = (const float*)d_in[3];
  const float* bq     = (const float*)d_in[4];
  const float* Wv     = (const float*)d_in[5];
  const float* bv     = (const float*)d_in[6];
  const float* Wo     = (const float*)d_in[7];
  const float* bo     = (const float*)d_in[8];
  const float* gammas = (const float*)d_in[9];
  const float* ln_g   = (const float*)d_in[10];
  const float* ln_b   = (const float*)d_in[11];
  float* out = (float*)d_out;

  const size_t NTD = 4ull * 1024 * 1024;  // B*T*D
  const size_t NDD = 1024ull * 1024;      // D*D

  unsigned short* qbf = (unsigned short*)d_ws;
  unsigned short* kbf = qbf + NTD;
  unsigned short* vbf = kbf + NTD;
  unsigned short* wqb = vbf + NTD;
  unsigned short* wvb = wqb + NDD;
  unsigned short* wob = wvb + NDD;
  unsigned short* Qp  = wob + NDD;   // (B,H,T,DK) bf16
  unsigned short* Kp  = Qp + NTD;    // (B,H,T,DK) bf16
  unsigned short* Vt  = Kp + NTD;    // (B,H,DK,T) bf16
  unsigned short* cat = Vt + NTD;    // (B,T,D) bf16
  float* attnf = (float*)(cat + NTD);

  dim3 blk(256);
  cvt_bf16<<<(int)(NTD / 1024), blk, 0, stream>>>(query,  qbf, (int)NTD);
  cvt_bf16<<<(int)(NTD / 1024), blk, 0, stream>>>(key,    kbf, (int)NTD);
  cvt_bf16<<<(int)(NTD / 1024), blk, 0, stream>>>(values, vbf, (int)NTD);
  cvt_bf16<<<(int)(NDD / 1024), blk, 0, stream>>>(Wq, wqb, (int)NDD);
  cvt_bf16<<<(int)(NDD / 1024), blk, 0, stream>>>(Wv, wvb, (int)NDD);
  cvt_bf16<<<(int)(NDD / 1024), blk, 0, stream>>>(Wo, wob, (int)NDD);

  dim3 ggrid(32, 8);  // 4096/128 x 1024/128
  gemm_wmma<<<ggrid, blk, 0, stream>>>(qbf, wqb, bq, Qp,   4096, 1024, 1024, 0);
  gemm_wmma<<<ggrid, blk, 0, stream>>>(kbf, wqb, bq, Kp,   4096, 1024, 1024, 0);  // kq_same
  gemm_wmma<<<ggrid, blk, 0, stream>>>(vbf, wvb, bv, Vt,   4096, 1024, 1024, 1);

  dim3 agrid(64, 64);  // T/16 query tiles x B*H heads
  attn_core<<<agrid, blk, 0, stream>>>(Qp, Kp, Vt, gammas, cat);

  gemm_wmma<<<ggrid, blk, 0, stream>>>(cat, wob, bo, attnf, 4096, 1024, 1024, 2);

  resid_ln<<<4096, blk, 0, stream>>>(query, attnf, ln_g, ln_b, out);
}
